// MixtureOfExperts_13675175870662
// MI455X (gfx1250) — compile-verified, matrix operand order checked by reading, not run on hardware
//
#include <hip/hip_runtime.h>
#include <hip/hip_bf16.h>

// ---------------- problem constants (from reference) ----------------
#define T_TOK 4096      // B*S tokens
#define D_DIM 1024
#define F_DIM 4096
#define N_EXP 8
#define BM    32        // tokens per expert-tile (2 WMMA M-tiles)
#define FC    128       // F chunk per outer iteration
#define KC    128       // K chunk for GEMM1 staging
#define NTHR  256       // 8 waves

typedef __attribute__((ext_vector_type(16))) __bf16 bf16x16;
typedef __attribute__((ext_vector_type(8)))  float  f32x8;
typedef int v4i __attribute__((ext_vector_type(4)));
typedef v4i __attribute__((address_space(1)))* gp_v4i;   // global (AS1) v4i32*
typedef v4i __attribute__((address_space(3)))* lp_v4i;   // LDS (AS3) v4i32*

// ---- async global->LDS copy (CDNA5 GLOBAL_LOAD_ASYNC_TO_LDS_B128) ----
#if __has_builtin(__builtin_amdgcn_global_load_async_to_lds_b128)
#define HAVE_ASYNC 1
#else
#define HAVE_ASYNC 0
#endif

static __device__ __forceinline__ void cp16(__hip_bfloat16* dst_lds,
                                            const __hip_bfloat16* src_g) {
#if HAVE_ASYNC
  __builtin_amdgcn_global_load_async_to_lds_b128((gp_v4i)src_g, (lp_v4i)dst_lds,
                                                 0, 0);
#else
  *(uint4*)dst_lds = *(const uint4*)src_g;
#endif
}

static __device__ __forceinline__ void wait_cp() {
#if HAVE_ASYNC
#if __has_builtin(__builtin_amdgcn_s_wait_asynccnt)
  __builtin_amdgcn_s_wait_asynccnt(0);
#else
  asm volatile("s_wait_asynccnt 0" ::: "memory");
#endif
#endif
}

static __device__ __forceinline__ bf16x16 load_frag(const __hip_bfloat16* lo,
                                                    const __hip_bfloat16* hi) {
  union { bf16x16 v; uint4 q[2]; } f;
  f.q[0] = *(const uint4*)lo;   // 8 bf16 = 16B  -> ds_load_b128
  f.q[1] = *(const uint4*)hi;
  return f.v;
}

static __device__ __forceinline__ f32x8 wmma_bf16(bf16x16 a, bf16x16 b, f32x8 c) {
  return __builtin_amdgcn_wmma_f32_16x16x32_bf16(false, a, false, b, (short)0, c,
                                                 false, false);
}

// ---------------- kernel 1: zero output + counters ----------------
__global__ void moe_zero_kernel(float* __restrict__ out, int n, int* __restrict__ cnt) {
  int i = blockIdx.x * blockDim.x + threadIdx.x;
  if (i < n) out[i] = 0.0f;
  if (blockIdx.x == 0 && threadIdx.x < N_EXP) cnt[threadIdx.x] = 0;
}

// ---------------- prep kernels: one-time bf16 convert / transpose ----------------
// After these, total weight footprint is 134 MB bf16 -> resident in 192 MB L2.
__global__ void moe_cvt_x_kernel(const float* __restrict__ x,
                                 __hip_bfloat16* __restrict__ xbf, int n) {
  int i = blockIdx.x * blockDim.x + threadIdx.x;
  if (i < n) xbf[i] = __float2bfloat16(x[i]);
}

// w1 (E,D,F) f32 -> w1t (E,F,D) bf16   (B-operand wants K=D contiguous)
__global__ void moe_tr_w1_kernel(const float* __restrict__ w1,
                                 __hip_bfloat16* __restrict__ w1t) {
  int i = blockIdx.x * blockDim.x + threadIdx.x;                 // over E*F*D
  int e = i >> 22;                                               // F*D = 2^22
  int r = i & ((1 << 22) - 1);
  int f = r >> 10, k = r & (D_DIM - 1);
  w1t[i] = __float2bfloat16(w1[((size_t)e * D_DIM + k) * F_DIM + f]);
}

// w2 (E,F,D) f32 -> w2t (E,D,F) bf16   (B-operand wants K=F contiguous)
__global__ void moe_tr_w2_kernel(const float* __restrict__ w2,
                                 __hip_bfloat16* __restrict__ w2t) {
  int i = blockIdx.x * blockDim.x + threadIdx.x;                 // over E*D*F
  int e = i >> 22;
  int r = i & ((1 << 22) - 1);
  int d = r >> 12, f = r & (F_DIM - 1);
  w2t[i] = __float2bfloat16(w2[((size_t)e * F_DIM + f) * D_DIM + d]);
}

// ---------------- kernel 2: gating (one wave32 per token) ----------------
__global__ void moe_gate_kernel(const float* __restrict__ x,
                                const float* __restrict__ gw,
                                int* __restrict__ cnt,
                                int* __restrict__ lists,
                                float* __restrict__ slists) {
  const int lane = threadIdx.x & 31;
  const int wave = threadIdx.x >> 5;
  const int t    = blockIdx.x * (NTHR / 32) + wave;

  float acc[N_EXP];
#pragma unroll
  for (int e = 0; e < N_EXP; ++e) acc[e] = 0.0f;

  const float* xr = x + (size_t)t * D_DIM;
  for (int i = lane; i < D_DIM; i += 32) {
    float xv = xr[i];
    const float* g = gw + (size_t)i * N_EXP;
#pragma unroll
    for (int e = 0; e < N_EXP; ++e) acc[e] += xv * g[e];
  }
#pragma unroll
  for (int e = 0; e < N_EXP; ++e)
    for (int off = 16; off > 0; off >>= 1) acc[e] += __shfl_down(acc[e], off, 32);

  if (lane == 0) {
    float mx = acc[0];
#pragma unroll
    for (int e = 1; e < N_EXP; ++e) mx = fmaxf(mx, acc[e]);
    float p[N_EXP];
#pragma unroll
    for (int e = 0; e < N_EXP; ++e) p[e] = __expf(acc[e] - mx);
    int i0 = 0;
#pragma unroll
    for (int e = 1; e < N_EXP; ++e) if (p[e] > p[i0]) i0 = e;
    int i1 = (i0 == 0) ? 1 : 0;
#pragma unroll
    for (int e = 0; e < N_EXP; ++e) if (e != i0 && p[e] > p[i1]) i1 = e;
    float s0 = p[i0], s1 = p[i1];
    float inv = 1.0f / (s0 + s1);       // softmax denominator cancels
    s0 *= inv; s1 *= inv;
    int p0 = atomicAdd(&cnt[i0], 1);
    lists[i0 * T_TOK + p0]  = t;
    slists[i0 * T_TOK + p0] = s0;
    int p1 = atomicAdd(&cnt[i1], 1);
    lists[i1 * T_TOK + p1]  = t;
    slists[i1 * T_TOK + p1] = s1;
  }
}

// ---------------- kernel 3: grouped expert FFN (WMMA bf16, async staging) ----------------
// block = 256 threads (8 waves); grid = (T/BM tiles, N_EXP experts)
// LDS: xs[32][1024] + w1s[128][128] (f-major) + w2s[1024][32] (d-major)
//      + hs[32][128] + tok/sc  = 168.2 KB  (<= 320 KB / WGP)
__global__ __launch_bounds__(NTHR)
void moe_expert_kernel(const __hip_bfloat16* __restrict__ xbf,
                       const __hip_bfloat16* __restrict__ w1t,
                       const float* __restrict__ b1,
                       const __hip_bfloat16* __restrict__ w2t,
                       const float* __restrict__ b2,
                       const int* __restrict__ cnt,
                       const int* __restrict__ lists,
                       const float* __restrict__ slists,
                       float* __restrict__ out) {
  extern __shared__ __align__(16) char smem_raw[];
  __hip_bfloat16* xs  = (__hip_bfloat16*)smem_raw;        // [32][1024]
  __hip_bfloat16* w1s = xs  + BM * D_DIM;                 // [FC][KC]  f-major
  __hip_bfloat16* w2s = w1s + FC * KC;                    // [1024][32] d-major
  __hip_bfloat16* hs  = w2s + D_DIM * 32;                 // [32][FC]
  int*   tokS = (int*)(hs + BM * FC);
  float* scS  = (float*)(tokS + BM);

  const int e     = blockIdx.y;
  const int tile  = blockIdx.x;
  const int count = cnt[e];
  if (tile * BM >= count) return;

  const int tid   = threadIdx.x;
  const int lane  = tid & 31;
  const int wave  = tid >> 5;
  const int nlane = lane & 15;
  const int kb    = (lane < 16) ? 0 : 8;     // per-lane K sub-run base
  const int mhi   = (lane < 16) ? 0 : 8;     // C/D-layout row offset

  if (tid < BM) {
    int idx = tile * BM + tid;
    int tkn = (idx < count) ? lists[e * T_TOK + idx] : -1;
    tokS[tid] = tkn;
    scS[tid]  = (idx < count) ? slists[e * T_TOK + idx] : 0.0f;
  }
  __syncthreads();

  // gathered x rows: contiguous bf16, 16B chunks (async -> LDS)
  for (int i = tid; i < BM * (D_DIM / 8); i += NTHR) {
    int m = i >> 7, c = i & 127;
    int tkn = tokS[m];
    const __hip_bfloat16* src =
        &xbf[(size_t)(tkn >= 0 ? tkn : 0) * D_DIM + c * 8];
    cp16(&xs[m * D_DIM + c * 8], src);
  }
  wait_cp();

  const __hip_bfloat16* w1te = w1t + (size_t)e * F_DIM * D_DIM;  // (F,D)
  const __hip_bfloat16* w2te = w2t + (size_t)e * D_DIM * F_DIM;  // (D,F)
  const float* b1e = b1 + e * F_DIM;
  const float* b2e = b2 + e * D_DIM;

  f32x8 yacc[16];
#pragma unroll
  for (int i = 0; i < 16; ++i) yacc[i] = (f32x8){0,0,0,0,0,0,0,0};

  for (int f0 = 0; f0 < F_DIM; f0 += FC) {
    // ---- GEMM1: h = x @ w1[:, f0:f0+FC]  (K = D in KC slabs) ----
    f32x8 hacc[2];
    hacc[0] = (f32x8){0,0,0,0,0,0,0,0};
    hacc[1] = (f32x8){0,0,0,0,0,0,0,0};
    for (int k0 = 0; k0 < D_DIM; k0 += KC) {
      __syncthreads();                       // prior consumers done
      if (tid == 0 && k0 + KC < D_DIM)
        __builtin_prefetch(&w1te[(size_t)f0 * D_DIM + k0 + KC], 0, 1);
      // w1t rows f0..f0+127, cols k0..k0+127 : contiguous 16B chunks
      for (int i = tid; i < FC * (KC / 8); i += NTHR) {
        int fl = i >> 4, c = i & 15;
        cp16(&w1s[fl * KC + c * 8],
             &w1te[(size_t)(f0 + fl) * D_DIM + k0 + c * 8]);
      }
      wait_cp();
      __syncthreads();                       // staging visible
#pragma unroll
      for (int kk = 0; kk < KC; kk += 32) {
        bf16x16 a0 = load_frag(&xs[nlane * D_DIM + k0 + kk + kb],
                               &xs[nlane * D_DIM + k0 + kk + kb + 16]);
        bf16x16 a1 = load_frag(&xs[(16 + nlane) * D_DIM + k0 + kk + kb],
                               &xs[(16 + nlane) * D_DIM + k0 + kk + kb + 16]);
        bf16x16 b  = load_frag(&w1s[(16 * wave + nlane) * KC + kk + kb],
                               &w1s[(16 * wave + nlane) * KC + kk + kb + 16]);
        hacc[0] = wmma_bf16(a0, b, hacc[0]);
        hacc[1] = wmma_bf16(a1, b, hacc[1]);
      }
    }
    // ---- bias + exact GELU -> hs (bf16) ----
    __syncthreads();
    {
      float bias = b1e[f0 + 16 * wave + nlane];
#pragma unroll
      for (int mt = 0; mt < 2; ++mt) {
#pragma unroll
        for (int j = 0; j < 8; ++j) {
          int m = 16 * mt + j + mhi;
          float v = hacc[mt][j] + bias;
          v = 0.5f * v * (1.0f + erff(v * 0.70710678f));
          hs[m * FC + 16 * wave + nlane] = __float2bfloat16(v);
        }
      }
    }
    // ---- GEMM2: y += h @ w2[f0:f0+FC, :]  (K = FC in 32-steps) ----
    for (int kk = 0; kk < FC; kk += 32) {
      __syncthreads();
      // w2t rows d=0..1023, cols f0+kk..+31 : 64B per row (4 x 16B chunks)
      for (int i = tid; i < D_DIM * 4; i += NTHR) {
        int d = i >> 2, c = i & 3;
        cp16(&w2s[d * 32 + c * 8],
             &w2te[(size_t)d * F_DIM + f0 + kk + c * 8]);
      }
      wait_cp();
      __syncthreads();
      bf16x16 a0 = load_frag(&hs[nlane * FC + kk + kb],
                             &hs[nlane * FC + kk + kb + 16]);
      bf16x16 a1 = load_frag(&hs[(16 + nlane) * FC + kk + kb],
                             &hs[(16 + nlane) * FC + kk + kb + 16]);
#pragma unroll
      for (int nt = 0; nt < 8; ++nt) {
        int dcol = 128 * wave + 16 * nt + nlane;
        bf16x16 b = load_frag(&w2s[dcol * 32 + kb],
                              &w2s[dcol * 32 + kb + 16]);
        yacc[nt]     = wmma_bf16(a0, b, yacc[nt]);
        yacc[8 + nt] = wmma_bf16(a1, b, yacc[8 + nt]);
      }
    }
  }

  // ---- epilogue: out[t, :] += score * (y + b2) ----
  __syncthreads();
#pragma unroll
  for (int nt = 0; nt < 8; ++nt) {
    int dcol = 128 * wave + 16 * nt + nlane;
    float bias = b2e[dcol];
#pragma unroll
    for (int mt = 0; mt < 2; ++mt) {
#pragma unroll
      for (int j = 0; j < 8; ++j) {
        int m = 16 * mt + j + mhi;
        int tkn = tokS[m];
        if (tkn >= 0)
          atomicAdd(&out[(size_t)tkn * D_DIM + dcol],
                    scS[m] * (yacc[mt * 8 + nt][j] + bias));
      }
    }
  }
}

// ---------------- launch ----------------
extern "C" void kernel_launch(void* const* d_in, const int* in_sizes, int n_in,
                              void* d_out, int out_size, void* d_ws, size_t ws_size,
                              hipStream_t stream) {
  const float* x   = (const float*)d_in[0];
  const float* gw  = (const float*)d_in[1];
  const float* w1  = (const float*)d_in[2];
  const float* b1  = (const float*)d_in[3];
  const float* w2  = (const float*)d_in[4];
  const float* b2  = (const float*)d_in[5];
  float* out = (float*)d_out;

  // workspace layout (bytes):
  //  [cnt 256] [lists E*T*4] [slists E*T*4] [xbf T*D*2] [w1t E*F*D*2] [w2t E*D*F*2]
  char* ws = (char*)d_ws;
  int*            cnt    = (int*)ws;
  int*            lists  = (int*)(ws + 256);
  float*          slists = (float*)(ws + 256 + N_EXP * T_TOK * 4);
  __hip_bfloat16* xbf    = (__hip_bfloat16*)(ws + 256 + 2 * N_EXP * T_TOK * 4);
  __hip_bfloat16* w1t    = xbf + (size_t)T_TOK * D_DIM;
  __hip_bfloat16* w2t    = w1t + (size_t)N_EXP * F_DIM * D_DIM;

  int n_out = T_TOK * D_DIM;
  moe_zero_kernel<<<(n_out + NTHR - 1) / NTHR, NTHR, 0, stream>>>(out, n_out, cnt);

  int n_x = T_TOK * D_DIM;
  moe_cvt_x_kernel<<<(n_x + NTHR - 1) / NTHR, NTHR, 0, stream>>>(x, xbf, n_x);
  int n_w = N_EXP * D_DIM * F_DIM;
  moe_tr_w1_kernel<<<n_w / NTHR, NTHR, 0, stream>>>(w1, w1t);
  moe_tr_w2_kernel<<<n_w / NTHR, NTHR, 0, stream>>>(w2, w2t);

  moe_gate_kernel<<<T_TOK / (NTHR / 32), NTHR, 0, stream>>>(x, gw, cnt, lists, slists);

  size_t smem = (size_t)(BM * D_DIM + FC * KC + D_DIM * 32 + BM * FC) *
                    sizeof(__hip_bfloat16) +
                BM * (sizeof(int) + sizeof(float));
  (void)hipFuncSetAttribute((const void*)moe_expert_kernel,
                            hipFuncAttributeMaxDynamicSharedMemorySize, (int)smem);
  dim3 grid(T_TOK / BM, N_EXP);
  moe_expert_kernel<<<grid, NTHR, smem, stream>>>(xbf, w1t, b1, w2t, b2, cnt,
                                                  lists, slists, out);
}
